// Attend_51410758533699
// MI455X (gfx1250) — matrix operand order repaired and checked by reading, not compiled, not run on hardware
//
#include <hip/hip_runtime.h>

typedef __attribute__((ext_vector_type(16))) _Float16 v16h;
typedef __attribute__((ext_vector_type(8)))  _Float16 v8h;
typedef __attribute__((ext_vector_type(8)))  float    v8f;

#define B_   4
#define H_   16
#define N_   2048
#define D_   64
#define KVT  64            // kv rows per tile (4 x 16 column chunks)
#define WAVES 4
#define QPW  32            // query rows per wave (2 x 16 A-tiles, B-frag reuse)
#define QPB  128           // query rows per block (4 waves x 32)

// LDS strides in halves (72 = 144B: 16B-aligned, bank-conflict friendly)
#define KSTR 72            // K tile: [64 kv][64 d]
#define VSTR 72            // V tile transposed: [64 d][64 kv]
#define PSTR 72            // P tile per wave: [32 row][64 col]

union V16 { v16h v; v8h h[2]; };

__device__ inline v8h cvt8(const float4 a, const float4 b, float s) {
  v8h r;
  r[0]=(_Float16)(a.x*s); r[1]=(_Float16)(a.y*s); r[2]=(_Float16)(a.z*s); r[3]=(_Float16)(a.w*s);
  r[4]=(_Float16)(b.x*s); r[5]=(_Float16)(b.y*s); r[6]=(_Float16)(b.z*s); r[7]=(_Float16)(b.w*s);
  return r;
}

// halves [0,8) and [16,24) from LDS -> v16h (two ds_load_b128)
__device__ inline v16h lds16(const _Float16* base) {
  V16 u;
  u.h[0] = *(const v8h*)(base);
  u.h[1] = *(const v8h*)(base + 16);
  return u.v;
}

// ---- row-max reduction within 16-lane groups ------------------------------
// v_permlane16_b32 butterfly (pure VALU, co-executes with the WMMA pipe and
// avoids ds_bpermute's shared-DScnt serialization). Nibble tables implement
// lane^1, lane^2, lane^4, lane^8 within each 16-lane row.
#if __has_builtin(__builtin_amdgcn_permlane16)
__device__ inline float plmax(float x, int lo, int hi_) {
  int xi = __float_as_int(x);
  int yi = __builtin_amdgcn_permlane16(xi, xi, lo, hi_, false, false);
  return fmaxf(x, __int_as_float(yi));
}
__device__ inline float rowmax16(float x) {
  x = plmax(x, (int)0x67452301u, (int)0xEFCDAB89u);  // xor 1
  x = plmax(x, (int)0x54761032u, (int)0xDCFE98BAu);  // xor 2
  x = plmax(x, (int)0x32107654u, (int)0xBA98FEDCu);  // xor 4
  x = plmax(x, (int)0xFEDCBA98u, (int)0x76543210u);  // xor 8
  return x;
}
#else
__device__ inline float rowmax16(float x) {
  x = fmaxf(x, __shfl_xor(x, 1));
  x = fmaxf(x, __shfl_xor(x, 2));
  x = fmaxf(x, __shfl_xor(x, 4));
  x = fmaxf(x, __shfl_xor(x, 8));
  return x;
}
#endif

#define WMMA(a, b, c) __builtin_amdgcn_wmma_f32_16x16x32_f16(false, a, false, b, (short)0, c, false, false)

__global__ __launch_bounds__(128) void
attn_flash_wmma(const float* __restrict__ q, const float* __restrict__ k,
                const float* __restrict__ v, float* __restrict__ out)
{
  __shared__ __align__(32) _Float16 kT[KVT * KSTR];          // 9216 B
  __shared__ __align__(32) _Float16 vT[D_ * VSTR];           // 9216 B
  __shared__ __align__(32) _Float16 pT[WAVES * 32 * PSTR];   // 18432 B

  const int tid    = threadIdx.x;
  const int lane   = tid & 31;
  const int wave   = tid >> 5;
  const int lane16 = lane & 15;
  const int hi     = lane >> 4;

  const int bh    = blockIdx.x;
  const int b     = bh / H_;
  const int h     = bh % H_;
  const int qBase = blockIdx.y * QPB + wave * QPW;

  const size_t headOff = (size_t)bh * N_ * D_;
  const float* qHead = q + headOff;
  const float* kHead = k + headOff;
  const float* vHead = v + headOff;

  // ---- Q fragments: 2 A-tiles per wave, pre-scaled by 1/sqrt(D)*log2(e) ----
  const float qscale = 0.125f * 1.44269504088896340736f;
  v16h qf[2][2];   // [qt][ks]
  #pragma unroll
  for (int qt = 0; qt < 2; ++qt) {
    const float* qRow = qHead + (size_t)(qBase + qt * 16 + lane16) * D_;
    #pragma unroll
    for (int ks = 0; ks < 2; ++ks) {
      const int d0 = ks * 32 + hi * 8;
      const float4* p0 = (const float4*)(qRow + d0);
      const float4* p1 = (const float4*)(qRow + d0 + 16);
      V16 u;
      u.h[0] = cvt8(p0[0], p0[1], qscale);
      u.h[1] = cvt8(p1[0], p1[1], qscale);
      qf[qt][ks] = u.v;
    }
  }

  // all-ones B fragment: P row-sums computed by the matrix engine
  v16h ones;
  #pragma unroll
  for (int i = 0; i < 16; ++i) ones[i] = (_Float16)1.0f;

  v8f o[2][4];
  float m[2][8], l[2][8];
  #pragma unroll
  for (int qt = 0; qt < 2; ++qt) {
    #pragma unroll
    for (int r = 0; r < 8; ++r) { m[qt][r] = -1e30f; l[qt][r] = 0.0f; }
    #pragma unroll
    for (int nd = 0; nd < 4; ++nd) o[qt][nd] = (v8f){};
  }

  _Float16* pMine = pT + wave * 32 * PSTR;

  const int nTiles = N_ / KVT;
  for (int t = 0; t < nTiles; ++t) {
    const int kv0 = t * KVT;
    __syncthreads();

    // ---- stage K[64][64] f32 -> kT f16 row-major --------------------------
    {
      const int row = tid >> 1;            // 0..63
      const int d0  = (tid & 1) << 5;      // 0 or 32
      const float4* ks = (const float4*)(kHead + (size_t)(kv0 + row) * D_ + d0);
      float4 a0 = ks[0], a1 = ks[1], a2 = ks[2], a3 = ks[3];
      float4 a4 = ks[4], a5 = ks[5], a6 = ks[6], a7 = ks[7];
      *(v8h*)(kT + row * KSTR + d0 +  0) = cvt8(a0, a1, 1.0f);
      *(v8h*)(kT + row * KSTR + d0 +  8) = cvt8(a2, a3, 1.0f);
      *(v8h*)(kT + row * KSTR + d0 + 16) = cvt8(a4, a5, 1.0f);
      *(v8h*)(kT + row * KSTR + d0 + 24) = cvt8(a6, a7, 1.0f);
      if (t + 1 < nTiles)
        __builtin_prefetch(kHead + (size_t)(kv0 + KVT + row) * D_ + d0, 0, 0);
    }
    // ---- stage V[64][64] f32 -> vT f16 transposed [d][kv] -----------------
    {
      const int col = tid & 63;            // d index
      const int kvh = tid >> 6;            // 0 or 1
      const float* vcol = vHead + (size_t)(kv0 + kvh * 32) * D_ + col;
      _Float16 tmp[32];
      #pragma unroll
      for (int i = 0; i < 32; ++i) tmp[i] = (_Float16)vcol[(size_t)i * D_];
      #pragma unroll
      for (int g = 0; g < 4; ++g) {
        v8h x;
        #pragma unroll
        for (int j = 0; j < 8; ++j) x[j] = tmp[g * 8 + j];
        *(v8h*)(vT + col * VSTR + kvh * 32 + g * 8) = x;
      }
      if (t + 1 < nTiles)
        __builtin_prefetch(vHead + (size_t)(kv0 + KVT + kvh * 32) * D_ + col, 0, 0);
    }
    __syncthreads();

    // ---- S = Q K^T : every B-frag feeds both Q tiles (A reuse) ------------
    v8f s[2][4];
    #pragma unroll
    for (int qt = 0; qt < 2; ++qt)
      #pragma unroll
      for (int c = 0; c < 4; ++c) s[qt][c] = (v8f){};
    #pragma unroll
    for (int ks = 0; ks < 2; ++ks) {
      const int dofs = ks * 32 + hi * 8;
      #pragma unroll
      for (int c = 0; c < 4; ++c) {
        v16h bk = lds16(kT + (c * 16 + lane16) * KSTR + dofs);
        s[0][c] = WMMA(qf[0][ks], bk, s[0][c]);
        s[1][c] = WMMA(qf[1][ks], bk, s[1][c]);
      }
    }

    // ---- online softmax per Q tile (exp2 domain) --------------------------
    v16h pa[2][2];   // A-frags of P, kept live for the PV phase
    #pragma unroll
    for (int qt = 0; qt < 2; ++qt) {
      float alpha[8];
      v8f p0, p1, p2, p3;
      #pragma unroll
      for (int r = 0; r < 8; ++r) {
        float mx = fmaxf(fmaxf(s[qt][0][r], s[qt][1][r]),
                         fmaxf(s[qt][2][r], s[qt][3][r]));
        mx = rowmax16(mx);
        const float mn = fmaxf(m[qt][r], mx);
        alpha[r] = exp2f(m[qt][r] - mn);
        p0[r] = exp2f(s[qt][0][r] - mn);
        p1[r] = exp2f(s[qt][1][r] - mn);
        p2[r] = exp2f(s[qt][2][r] - mn);
        p3[r] = exp2f(s[qt][3][r] - mn);
        m[qt][r] = mn;
      }
      #pragma unroll
      for (int r = 0; r < 8; ++r) {
        o[qt][0][r] *= alpha[r]; o[qt][1][r] *= alpha[r];
        o[qt][2][r] *= alpha[r]; o[qt][3][r] *= alpha[r];
      }
      // P: C-layout f32 -> A-layout f16 via wave-private LDS patch
      #pragma unroll
      for (int r = 0; r < 8; ++r) {
        _Float16* pr = pMine + (qt * 16 + r + 8 * hi) * PSTR + lane16;
        pr[ 0] = (_Float16)p0[r];
        pr[16] = (_Float16)p1[r];
        pr[32] = (_Float16)p2[r];
        pr[48] = (_Float16)p3[r];
      }
      {
        const _Float16* pm = pMine + (qt * 16 + lane16) * PSTR + hi * 8;
        V16 u0, u1;
        u0.h[0] = *(const v8h*)(pm);        u0.h[1] = *(const v8h*)(pm + 16);
        u1.h[0] = *(const v8h*)(pm + 32);   u1.h[1] = *(const v8h*)(pm + 48);
        pa[qt][0] = u0.v; pa[qt][1] = u1.v;
      }
      // P row-sums on the matrix engine (broadcast across lanes in C layout)
      v8f rs = {};
      rs = WMMA(pa[qt][0], ones, rs);
      rs = WMMA(pa[qt][1], ones, rs);
      #pragma unroll
      for (int r = 0; r < 8; ++r) l[qt][r] = l[qt][r] * alpha[r] + rs[r];
    }

    // ---- O += P V : every V B-frag feeds both Q tiles ---------------------
    {
      const int kvo = hi * 8;
      #pragma unroll
      for (int nd = 0; nd < 4; ++nd) {
        v16h bv0 = lds16(vT + (nd * 16 + lane16) * VSTR + kvo);
        v16h bv1 = lds16(vT + (nd * 16 + lane16) * VSTR + 32 + kvo);
        o[0][nd] = WMMA(pa[0][0], bv0, o[0][nd]);
        o[0][nd] = WMMA(pa[0][1], bv1, o[0][nd]);
        o[1][nd] = WMMA(pa[1][0], bv0, o[1][nd]);
        o[1][nd] = WMMA(pa[1][1], bv1, o[1][nd]);
      }
    }
  }

  // ---- epilogue: O /= l, store to out[b][n][h*D + d] ----------------------
  #pragma unroll
  for (int qt = 0; qt < 2; ++qt)
    #pragma unroll
    for (int r = 0; r < 8; ++r) {
      const float inv = 1.0f / l[qt][r];
      const int row = qBase + qt * 16 + r + 8 * hi;
      float* orow = out + ((size_t)b * N_ + row) * (H_ * D_) + h * D_;
      orow[ 0 + lane16] = o[qt][0][r] * inv;
      orow[16 + lane16] = o[qt][1][r] * inv;
      orow[32 + lane16] = o[qt][2][r] * inv;
      orow[48 + lane16] = o[qt][3][r] * inv;
    }
}

extern "C" void kernel_launch(void* const* d_in, const int* in_sizes, int n_in,
                              void* d_out, int out_size, void* d_ws, size_t ws_size,
                              hipStream_t stream) {
  (void)in_sizes; (void)n_in; (void)out_size; (void)d_ws; (void)ws_size;
  const float* q = (const float*)d_in[0];
  const float* k = (const float*)d_in[1];
  const float* v = (const float*)d_in[2];
  float* out = (float*)d_out;

  dim3 grid(B_ * H_, N_ / QPB);   // (64, 16)
  dim3 block(128);                // 4 wave32 waves
  attn_flash_wmma<<<grid, block, 0, stream>>>(q, k, v, out);
}
